// GatedBridgeInjector_91199335563488
// MI455X (gfx1250) — matrix operand order repaired and checked by reading, not compiled, not run on hardware
//
#include <hip/hip_runtime.h>
#include <math.h>

// ---------------------------------------------------------------------------
// GatedBridgeInjector for MI455X (gfx1250, wave32, WMMA bf16 16x16x32).
// Compute-bound (~110 GFLOP vs ~0.2 GB traffic) -> all matmuls on
// v_wmma_f32_16x16x32_bf16 with f32 accumulation.
// Round 3: QKV GEMMs emit bf16 directly so the flash-attention inner loop
// does zero f32->bf16 converts and half the load bytes. TDM weight tiles +
// prefetch + transposed-score softmax retained from round 2.
// ---------------------------------------------------------------------------

#define Bsz  4
#define Ssz  2048
#define Hsz  768
#define NHsz 12
#define HDsz 64
#define ROWS (Bsz * Ssz)          // 8192 token rows
#define HH   (Hsz * Hsz)

typedef __attribute__((ext_vector_type(16))) __bf16 v16bf;
typedef __attribute__((ext_vector_type(4)))  __bf16 v4bf;
typedef __attribute__((ext_vector_type(8)))  float  v8f;
typedef __attribute__((ext_vector_type(4)))  unsigned int v4u;
typedef __attribute__((ext_vector_type(8)))  int v8i;
typedef __attribute__((ext_vector_type(4)))  int v4i;

// A-matrix (16x32, 16-bit) K index for element e, lane-half h (ISA 7.12.2):
// lanes 0-15: V0..V3 = K0..7, V4..V7 = K16..23 ; lanes 16-31: K8..15, K24..31
__device__ __forceinline__ int a_kidx(int e, int half) {
  return ((e & 8) << 1) + (half << 3) + (e & 7);
}

// LDS byte offset of a __shared__ object: flat shared-aperture addresses carry
// the LDS offset in the low 32 bits (ISA 10.2 aperture mapping).
__device__ __forceinline__ unsigned lds_off(const void* p) {
  return (unsigned)(unsigned long long)p;
}

// Issue a 2D TDM load: tile_d0 x tile_d1 elements of 2 bytes, row stride
// `stride` elements, from `gaddr` into LDS at byte offset `laddr`.
__device__ __forceinline__ void tdm_load_2d_b16(unsigned laddr, const void* gaddr,
                                                unsigned tile_d0, unsigned tile_d1,
                                                unsigned stride) {
  unsigned long long ga = (unsigned long long)gaddr;
  v4u g0;
  g0[0] = 1u;                                    // count=1 (valid), user mode
  g0[1] = laddr;                                 // lds_addr
  g0[2] = (unsigned)(ga & 0xFFFFFFFFu);          // global_addr[31:0]
  g0[3] = (unsigned)((ga >> 32) & 0x1FFFFFFu)    // global_addr[56:32]
          | (2u << 30);                          // type = 2 (image)
  v8i g1;
  g1[0] = (int)(1u << 16);                       // data_size = 1 -> 2 bytes
  g1[1] = (int)((stride & 0xFFFFu) << 16);       // tensor_dim0[15:0] (use stride)
  g1[2] = (int)(((stride >> 16) & 0xFFFFu)       // tensor_dim0[31:16]
          | ((65535u & 0xFFFFu) << 16));         // tensor_dim1[15:0] (large)
  g1[3] = (int)((tile_d0 & 0xFFFFu) << 16);      // tensor_dim1[31:16]=0, tile_dim0
  g1[4] = (int)(tile_d1 & 0xFFFFu);              // tile_dim1 ; tile_dim2 = 0
  g1[5] = (int)stride;                           // tensor_dim0_stride[31:0]
  g1[6] = 0;                                     // stride[47:32]=0, dim1_stride lo
  g1[7] = 0;
  v4i z4 = {};
#if defined(__clang_major__) && (__clang_major__ >= 23)
  v8i z8 = {};
  __builtin_amdgcn_tensor_load_to_lds(g0, g1, z4, z4, z8, 0);
#else
  __builtin_amdgcn_tensor_load_to_lds(g0, g1, z4, z4, 0);
#endif
}

// ---------------- weight f32 -> bf16 conversion (once) ----------------------
__global__ __launch_bounds__(256) void wconv_kernel(
    const float* __restrict__ w0, const float* __restrict__ w1,
    const float* __restrict__ w2, const float* __restrict__ w3,
    const float* __restrict__ w4, const float* __restrict__ w5,
    __bf16* __restrict__ dst) {
  int i = blockIdx.x * blockDim.x + threadIdx.x;
  if (i >= 6 * HH) return;
  int m = i / HH, r = i % HH;
  const float* src = (m == 0) ? w0 : (m == 1) ? w1 : (m == 2) ? w2
                    : (m == 3) ? w3 : (m == 4) ? w4 : w5;
  dst[i] = (__bf16)src[r];
}

// ---------------- depthwise conv1d (k=3, zero pad) --------------------------
__global__ __launch_bounds__(256) void dwconv_kernel(
    const float* __restrict__ ps, const float* __restrict__ w,
    const float* __restrict__ bias, float* __restrict__ out) {
  int i = blockIdx.x * blockDim.x + threadIdx.x;
  if (i >= ROWS * Hsz) return;
  int h = i % Hsz;
  int r = i / Hsz;
  int s = r % Ssz;
  int b = r / Ssz;
  const float* base = ps + (size_t)b * Ssz * Hsz + h;
  float w0 = w[h * 3 + 0], w1 = w[h * 3 + 1], w2 = w[h * 3 + 2];
  float acc = bias[h] + base[(size_t)s * Hsz] * w1;
  if (s > 0)       acc += base[(size_t)(s - 1) * Hsz] * w0;
  if (s < Ssz - 1) acc += base[(size_t)(s + 1) * Hsz] * w2;
  out[i] = acc;
}

// ---------------- WMMA GEMM: out[r,o] = act(sum_h A[r,h]*W[o,h] + bias[o]) --
// A: f32 [ROWS,H] (converted to bf16 through LDS), W: bf16 [H,H] row-major,
// fetched per K-step by the Tensor Data Mover. Block = 256 threads (8 waves),
// block tile 128x64, each wave 32x32.
template <int EPI, bool OBF>  // EPI: 0 = none, 1 = exact GELU; OBF: bf16 output
__global__ __launch_bounds__(256) void gemm_kernel(
    const float* __restrict__ A, const __bf16* __restrict__ W,
    const float* __restrict__ bias, void* __restrict__ outv, float scale) {
  __shared__ __align__(16) __bf16 As[128 * 32];
  __shared__ __align__(16) __bf16 Bs[64 * 32];
  const int t    = threadIdx.x;
  const int lane = t & 31;
  const int wv   = t >> 5;
  const int wm   = wv >> 1, wn = wv & 1;
  const int half = lane >> 4, lm = lane & 15;
  const int row0 = blockIdx.y * 128;
  const int col0 = blockIdx.x * 64;

  v8f acc00 = {}, acc01 = {}, acc10 = {}, acc11 = {};

  for (int k0 = 0; k0 < Hsz; k0 += 32) {
    // coalesced global loads of the f32 A tile into registers
    float4 av[4];
#pragma unroll
    for (int i = 0; i < 4; i++) {
      int flat4 = t + i * 256;          // 1024 float4 = 128x32 f32
      int row = flat4 >> 3, c4 = flat4 & 7;
      av[i] = *(const float4*)(A + (size_t)(row0 + row) * Hsz + k0 + c4 * 4);
      if (i == 0 && k0 + 32 < Hsz)      // prefetch next A tile (global_prefetch_b8)
        __builtin_prefetch(A + (size_t)(row0 + row) * Hsz + k0 + 32 + c4 * 4, 0, 1);
    }
    __syncthreads();                    // previous iteration's LDS reads done
    // weight tile 64x32 bf16 via Tensor Data Mover (wave 0 issues the DMA)
    if (wv == 0) {
      tdm_load_2d_b16(lds_off(Bs), W + (size_t)col0 * Hsz + k0, 32, 64, Hsz);
    }
#pragma unroll
    for (int i = 0; i < 4; i++) {
      int flat4 = t + i * 256;
      int row = flat4 >> 3, c4 = flat4 & 7;
      v4bf p;
      p[0] = (__bf16)av[i].x; p[1] = (__bf16)av[i].y;
      p[2] = (__bf16)av[i].z; p[3] = (__bf16)av[i].w;
      *(v4bf*)(As + row * 32 + c4 * 4) = p;
    }
    if (wv == 0) __builtin_amdgcn_s_wait_tensorcnt(0);
    __syncthreads();                    // tiles visible

    v16bf a0, a1, b0, b1;
#pragma unroll
    for (int e = 0; e < 16; e++) {
      int k = a_kidx(e, half);
      a0[e] = As[(wm * 32 + lm) * 32 + k];
      a1[e] = As[(wm * 32 + 16 + lm) * 32 + k];
    }
#pragma unroll
    for (int e = 0; e < 16; e++) {
      int k = a_kidx(e, half);          // B-fragment: lane = N, k in A-table order
      b0[e] = Bs[(wn * 32 + lm) * 32 + k];
      b1[e] = Bs[(wn * 32 + 16 + lm) * 32 + k];
    }
    acc00 = __builtin_amdgcn_wmma_f32_16x16x32_bf16(false, a0, false, b0, (short)0, acc00, false, false);
    acc01 = __builtin_amdgcn_wmma_f32_16x16x32_bf16(false, a0, false, b1, (short)0, acc01, false, false);
    acc10 = __builtin_amdgcn_wmma_f32_16x16x32_bf16(false, a1, false, b0, (short)0, acc10, false, false);
    acc11 = __builtin_amdgcn_wmma_f32_16x16x32_bf16(false, a1, false, b1, (short)0, acc11, false, false);
  }

  // epilogue (C layout: m = 8*half + j, n = lane&15)
#pragma unroll
  for (int mt = 0; mt < 2; mt++) {
#pragma unroll
    for (int nt = 0; nt < 2; nt++) {
      v8f acc = (mt == 0) ? ((nt == 0) ? acc00 : acc01)
                          : ((nt == 0) ? acc10 : acc11);
      int col = col0 + wn * 32 + nt * 16 + lm;
      float bb = bias[col];
#pragma unroll
      for (int j = 0; j < 8; j++) {
        int row = row0 + wm * 32 + mt * 16 + half * 8 + j;
        float v = acc[j] + bb;
        if (EPI == 1) v = 0.5f * v * (1.0f + erff(v * 0.70710678118654752f));
        v *= scale;
        if (OBF) ((__bf16*)outv)[(size_t)row * Hsz + col] = (__bf16)v;
        else     ((float*)outv)[(size_t)row * Hsz + col] = v;
      }
    }
  }
}

// ---------------- LayerNorm over H=768 (one block per row) ------------------
__global__ __launch_bounds__(256) void ln_kernel(
    const float* __restrict__ in, const float* __restrict__ g,
    const float* __restrict__ bb, float* __restrict__ out) {
  __shared__ float red[256];
  int r = blockIdx.x, t = threadIdx.x;
  const float* x = in + (size_t)r * Hsz;
  float v0 = x[t], v1 = x[t + 256], v2 = x[t + 512];
  red[t] = v0 + v1 + v2;
  __syncthreads();
  for (int o = 128; o > 0; o >>= 1) { if (t < o) red[t] += red[t + o]; __syncthreads(); }
  float mean = red[0] * (1.0f / Hsz);
  __syncthreads();
  float d0 = v0 - mean, d1 = v1 - mean, d2 = v2 - mean;
  red[t] = d0 * d0 + d1 * d1 + d2 * d2;
  __syncthreads();
  for (int o = 128; o > 0; o >>= 1) { if (t < o) red[t] += red[t + o]; __syncthreads(); }
  float rstd = rsqrtf(red[0] * (1.0f / Hsz) + 1e-5f);
  float* y = out + (size_t)r * Hsz;
  y[t]       = d0 * rstd * g[t]       + bb[t];
  y[t + 256] = d1 * rstd * g[t + 256] + bb[t + 256];
  y[t + 512] = d2 * rstd * g[t + 512] + bb[t + 512];
}

// ---------------- Flash attention: one wave per (b, head, 16-query tile) ----
// Q/K/V arrive as bf16 (rounded once in the QKV GEMM epilogue), so fragment
// loads are raw bf16 with no converts. Scores are computed TRANSPOSED
// (S^T = K x Q^T): queries land on lanes, so softmax max/sum are in-lane
// reductions + one xor-16 shuffle, and the S^T C-layout is bit-identical to
// the A-fragment layout needed for P x V (register-only repack, no LDS).
// q is pre-scaled by 1/sqrt(HD).
__global__ __launch_bounds__(32) void attn_kernel(
    const __bf16* __restrict__ q, const __bf16* __restrict__ k,
    const __bf16* __restrict__ v, float* __restrict__ out) {
  const int lane = threadIdx.x & 31;
  const int half = lane >> 4, lm = lane & 15;
  int tile = blockIdx.x;
  int qt   = tile & 127;                 // S/16 = 128 query tiles
  int head = (tile >> 7) % NHsz;
  int b    = tile / (128 * NHsz);
  const size_t base = (size_t)b * Ssz * Hsz;
  const __bf16* qp = q + base + head * HDsz;
  const __bf16* kp = k + base + head * HDsz;
  const __bf16* vp = v + base + head * HDsz;
  const int h0 = head * HDsz;
  const int q0 = qt * 16;

  // Q as B-fragments (columns = queries, k = head-dim chunk)
  v16bf qb0, qb1;
#pragma unroll
  for (int e = 0; e < 16; e++) {
    int d = a_kidx(e, half);
    qb0[e] = qp[(size_t)(q0 + lm) * Hsz + d];
    qb1[e] = qp[(size_t)(q0 + lm) * Hsz + 32 + d];
  }

  v8f o0 = {}, o1 = {}, o2 = {}, o3 = {};
  float mi = -3.0e38f, li = 0.0f;        // per-lane stats for query (q0 + lm)

  for (int key0 = 0; key0 < Ssz; key0 += 32) {
    // K as A-fragments (rows = keys)
    v16bf ka00, ka01, ka10, ka11;
#pragma unroll
    for (int e = 0; e < 16; e++) {
      int kk = a_kidx(e, half);
      size_t r0 = (size_t)(key0 + lm) * Hsz;
      size_t r1 = (size_t)(key0 + 16 + lm) * Hsz;
      ka00[e] = kp[r0 + kk];
      ka01[e] = kp[r0 + 32 + kk];
      ka10[e] = kp[r1 + kk];
      ka11[e] = kp[r1 + 32 + kk];
    }
    v8f s0 = {}, s1 = {};
    s0 = __builtin_amdgcn_wmma_f32_16x16x32_bf16(false, ka00, false, qb0, (short)0, s0, false, false);
    s0 = __builtin_amdgcn_wmma_f32_16x16x32_bf16(false, ka01, false, qb1, (short)0, s0, false, false);
    s1 = __builtin_amdgcn_wmma_f32_16x16x32_bf16(false, ka10, false, qb0, (short)0, s1, false, false);
    s1 = __builtin_amdgcn_wmma_f32_16x16x32_bf16(false, ka11, false, qb1, (short)0, s1, false, false);
    // s0[j] = S[key = 8*half+j][query = lm], s1 = keys +16

    // online softmax: in-lane max/sum over 16 keys + one xor-16 hop
    float mx = fmaxf(s0[0], s1[0]);
#pragma unroll
    for (int j = 1; j < 8; j++) mx = fmaxf(mx, fmaxf(s0[j], s1[j]));
    mx = fmaxf(mx, __shfl_xor(mx, 16, 32));
    float mnew  = fmaxf(mi, mx);
    float alpha = __expf(mi - mnew);
    float rs = 0.0f;
#pragma unroll
    for (int j = 0; j < 8; j++) {
      s0[j] = __expf(s0[j] - mnew);
      s1[j] = __expf(s1[j] - mnew);
      rs += s0[j] + s1[j];
    }
    rs += __shfl_xor(rs, 16, 32);
    li = li * alpha + rs;
    mi = mnew;

    // P^T C-layout == P A-fragment layout: pure register repack
    v16bf pa;
#pragma unroll
    for (int e = 0; e < 8; e++) {
      pa[e]     = (__bf16)s0[e];
      pa[e + 8] = (__bf16)s1[e];
    }

    // rescale accumulators: alpha for O-row m = 8*half+j lives in lane lm = m
#pragma unroll
    for (int j = 0; j < 8; j++) {
      float aj = __shfl(alpha, half * 24 + j, 32);
      o0[j] *= aj; o1[j] *= aj; o2[j] *= aj; o3[j] *= aj;
    }

    // V as B-fragments (k = key, columns = head-dim)
    v16bf vb0, vb1, vb2, vb3;
#pragma unroll
    for (int e = 0; e < 16; e++) {
      int kk = a_kidx(e, half);
      size_t rb = (size_t)(key0 + kk) * Hsz + lm;
      vb0[e] = vp[rb];
      vb1[e] = vp[rb + 16];
      vb2[e] = vp[rb + 32];
      vb3[e] = vp[rb + 48];
    }
    o0 = __builtin_amdgcn_wmma_f32_16x16x32_bf16(false, pa, false, vb0, (short)0, o0, false, false);
    o1 = __builtin_amdgcn_wmma_f32_16x16x32_bf16(false, pa, false, vb1, (short)0, o1, false, false);
    o2 = __builtin_amdgcn_wmma_f32_16x16x32_bf16(false, pa, false, vb2, (short)0, o2, false, false);
    o3 = __builtin_amdgcn_wmma_f32_16x16x32_bf16(false, pa, false, vb3, (short)0, o3, false, false);
  }

#pragma unroll
  for (int j = 0; j < 8; j++) {
    float lj = __shfl(li, half * 24 + j, 32);
    float inv = 1.0f / lj;
    size_t row = (size_t)(q0 + half * 8 + j);
    float* op = out + base + row * Hsz + h0 + lm;
    op[0]  = o0[j] * inv;
    op[16] = o1[j] * inv;
    op[32] = o2[j] * inv;
    op[48] = o3[j] * inv;
  }
}

// ---------------- gate + final injection + per-row aux partials -------------
__global__ __launch_bounds__(256) void gate_kernel(
    const float* __restrict__ x, const float* __restrict__ br,
    const float* __restrict__ inj, const unsigned char* __restrict__ mask,
    const float* __restrict__ Wg, const float* __restrict__ bg,
    float* __restrict__ out, float* __restrict__ auxrow) {
  __shared__ float red[256];
  __shared__ float red2[256];
  int r = blockIdx.x, t = threadIdx.x;
  int b = r / Ssz;
  const float* xr  = x  + (size_t)r * Hsz;
  const float* brr = br + (size_t)r * Hsz;
  float dot = 0.0f, aux = 0.0f;
#pragma unroll
  for (int i = 0; i < 3; i++) {
    int h = t + i * 256;
    float xv = xr[h], bv = brr[h];
    dot += xv * Wg[h] + bv * Wg[Hsz + h];
    float d = bv - xv;
    aux += d * d;
  }
  red[t] = dot; red2[t] = aux;
  __syncthreads();
  for (int o = 128; o > 0; o >>= 1) {
    if (t < o) { red[t] += red[t + o]; red2[t] += red2[t + o]; }
    __syncthreads();
  }
  float mf = mask[b] ? 1.0f : 0.0f;
  float gate = 1.0f / (1.0f + __expf(-(red[0] + bg[0])));
  if (t == 0) auxrow[r] = mf * red2[0];
  const float* ir = inj + (size_t)r * Hsz;
  float* orow = out + (size_t)r * Hsz;
#pragma unroll
  for (int i = 0; i < 3; i++) {
    int h = t + i * 256;
    orow[h] = xr[h] + mf * gate * tanhf(ir[h]);
  }
}

// ---------------- deterministic aux reduction (no float atomics) ------------
__global__ __launch_bounds__(256) void auxreduce_kernel(
    const float* __restrict__ auxrow, const unsigned char* __restrict__ mask,
    float* __restrict__ aux_out) {
  __shared__ float red[256];
  int t = threadIdx.x;
  float s = 0.0f;
  for (int i = t; i < ROWS; i += 256) s += auxrow[i];
  red[t] = s;
  __syncthreads();
  for (int o = 128; o > 0; o >>= 1) { if (t < o) red[t] += red[t + o]; __syncthreads(); }
  if (t == 0) {
    int nm = (mask[0] ? 1 : 0) + (mask[1] ? 1 : 0) + (mask[2] ? 1 : 0) + (mask[3] ? 1 : 0);
    if (nm < 1) nm = 1;
    aux_out[0] = red[0] / ((float)nm * (float)(Ssz * Hsz));
  }
}

// ---------------------------------------------------------------------------
extern "C" void kernel_launch(void* const* d_in, const int* in_sizes, int n_in,
                              void* d_out, int out_size, void* d_ws, size_t ws_size,
                              hipStream_t stream) {
  (void)in_sizes; (void)n_in; (void)out_size; (void)ws_size;
  const float* x     = (const float*)d_in[0];
  const float* prev  = (const float*)d_in[1];
  const unsigned char* mask = (const unsigned char*)d_in[2];  // jax bool = 1 byte
  const float* cdw_w = (const float*)d_in[3];
  const float* cdw_b = (const float*)d_in[4];
  const float* cpw_w = (const float*)d_in[5];
  const float* cpw_b = (const float*)d_in[6];
  const float* cn_g  = (const float*)d_in[7];
  const float* cn_b  = (const float*)d_in[8];
  const float* Wq = (const float*)d_in[9];   const float* bq = (const float*)d_in[10];
  const float* Wk = (const float*)d_in[11];  const float* bk = (const float*)d_in[12];
  const float* Wv = (const float*)d_in[13];  const float* bv = (const float*)d_in[14];
  const float* Wo = (const float*)d_in[15];  const float* bo = (const float*)d_in[16];
  const float* pn_g = (const float*)d_in[17]; const float* pn_b = (const float*)d_in[18];
  const float* Wpost = (const float*)d_in[19]; const float* bpost = (const float*)d_in[20];
  const float* Wg = (const float*)d_in[21];  const float* bg = (const float*)d_in[22];

  // workspace carve-out
  char* ws = (char*)d_ws;
  size_t off = 0;
  auto carve = [&](size_t bytes) {
    void* p = ws + off;
    off += (bytes + 255) & ~(size_t)255;
    return p;
  };
  __bf16* WB = (__bf16*)carve((size_t)6 * HH * sizeof(__bf16));
  float*  T0 = (float*)carve((size_t)ROWS * Hsz * sizeof(float));   // dw -> ao -> pn
  float*  T1 = (float*)carve((size_t)ROWS * Hsz * sizeof(float));   // pw -> attnWo -> inj
  float*  T2 = (float*)carve((size_t)ROWS * Hsz * sizeof(float));   // bridged (persist)
  __bf16* Q8 = (__bf16*)carve((size_t)ROWS * Hsz * sizeof(__bf16)); // q bf16 -> auxrow
  __bf16* K8 = (__bf16*)carve((size_t)ROWS * Hsz * sizeof(__bf16)); // k bf16
  __bf16* V8 = (__bf16*)carve((size_t)ROWS * Hsz * sizeof(__bf16)); // v bf16
  __bf16* WBpw   = WB + 0 * (size_t)HH;
  __bf16* WBq    = WB + 1 * (size_t)HH;
  __bf16* WBk    = WB + 2 * (size_t)HH;
  __bf16* WBv    = WB + 3 * (size_t)HH;
  __bf16* WBo    = WB + 4 * (size_t)HH;
  __bf16* WBpost = WB + 5 * (size_t)HH;

  float* outm = (float*)d_out;
  float* auxp = outm + (size_t)ROWS * Hsz;

  dim3 ggrid(Hsz / 64, ROWS / 128);  // (12, 64)

  // 1) weights -> bf16 (stay L2-resident: ~7 MB of 192 MB)
  wconv_kernel<<<(6 * HH + 255) / 256, 256, 0, stream>>>(cpw_w, Wq, Wk, Wv, Wo, Wpost, WB);
  // 2) depthwise conv -> T0
  dwconv_kernel<<<(ROWS * Hsz + 255) / 256, 256, 0, stream>>>(prev, cdw_w, cdw_b, T0);
  // 3) pointwise conv GEMM + exact GELU -> T1 (f32)
  gemm_kernel<1, false><<<ggrid, 256, 0, stream>>>(T0, WBpw, cpw_b, T1, 1.0f);
  // 4) LayerNorm(cn) -> T2 = bridged (f32, persists to the gate/aux stage)
  ln_kernel<<<ROWS, 256, 0, stream>>>(T1, cn_g, cn_b, T2);
  // 5) Q (scaled by 1/sqrt(64)), K, V projections -> bf16 directly
  gemm_kernel<0, true><<<ggrid, 256, 0, stream>>>(x,  WBq, bq, Q8, 0.125f);
  gemm_kernel<0, true><<<ggrid, 256, 0, stream>>>(T2, WBk, bk, K8, 1.0f);
  gemm_kernel<0, true><<<ggrid, 256, 0, stream>>>(T2, WBv, bv, V8, 1.0f);
  // 6) flash attention -> T0 (f32)
  attn_kernel<<<Bsz * NHsz * (Ssz / 16), 32, 0, stream>>>(Q8, K8, V8, T0);
  // 7) output projection Wo -> T1 (f32)
  gemm_kernel<0, false><<<ggrid, 256, 0, stream>>>(T0, WBo, bo, T1, 1.0f);
  // 8) LayerNorm(pn) -> T0
  ln_kernel<<<ROWS, 256, 0, stream>>>(T1, pn_g, pn_b, T0);
  // 9) Wpost GEMM -> T1 = inj (f32)
  gemm_kernel<0, false><<<ggrid, 256, 0, stream>>>(T0, WBpost, bpost, T1, 1.0f);
  // 10) gate + inject + aux partials -> out; per-row partials into dead Q8
  gate_kernel<<<ROWS, 256, 0, stream>>>(x, T2, T1, mask, Wg, bg, outm, (float*)Q8);
  // 11) deterministic aux scalar
  auxreduce_kernel<<<1, 256, 0, stream>>>((float*)Q8, mask, auxp);
}
